// MLAAttention_14594298871884
// MI455X (gfx1250) — matrix operand order repaired and checked by reading, not compiled, not run on hardware
//
#include <hip/hip_runtime.h>
#include <hip/hip_bf16.h>

// ---------------------------------------------------------------------------
// MLA attention forward for MI455X (gfx1250): bf16 WMMA everywhere,
// async global->LDS staging, wave32 flash attention.
// ---------------------------------------------------------------------------

#define USE_ASYNC 1

typedef __bf16 bf16_t;
typedef __attribute__((ext_vector_type(16))) __bf16        v16bf;
typedef __attribute__((ext_vector_type(8)))  float         v8f;
typedef __attribute__((ext_vector_type(8)))  unsigned int  v8u;

#define DEV __device__ __forceinline__

constexpr int  Bb   = 2;
constexpr int  Ss   = 2048;
constexpr int  Dd   = 2048;
constexpr int  Hh   = 16;
constexpr int  HDd  = 128;
constexpr int  RDd  = 64;
constexpr int  NDd  = 64;   // HD - RD
constexpr int  KVRr = 512;
constexpr int  QRr  = 1024;
constexpr int  Tt   = Bb * Ss;                 // 4096 tokens
constexpr float SCALE = 0.08838834764831845f;  // 1/sqrt(128)
constexpr float NEGF  = -1e30f;
constexpr float LOGB_OVER_HALF = 0.28782313662425575f; // ln(10000)/32

DEV v16bf as_v16bf(v8u u) { return __builtin_bit_cast(v16bf, u); }

DEV v8f wmma_bf16(v16bf a, v16bf b, v8f c) {
  // D = A(16x32) * B(32x16) + C, f32 accumulate
  return __builtin_amdgcn_wmma_f32_16x16x32_bf16(
      /*neg_a=*/false, a, /*neg_b=*/false, b,
      /*c_mod=*/(short)0, c, /*reuse_a=*/false, /*reuse_b=*/false);
}

// A-matrix (16x32, 16-bit) dword index within a 32-elem K row, per ISA layout:
// VGPR j<4 : K = 2j + 8h ; VGPR j>=4 : K = 16 + 2(j-4) + 8h  (h = lane/16)
DEV int a_dw(int j, int h) { return (j < 4 ? j : j + 4) + 4 * h; }
// B-matrix (32x16, 16-bit) dword index within a 32-elem K column (stored [n][k]):
// VGPR j<4 : K = 2j + 8h ; VGPR j>=4 : K = 16 + 2(j-4) + 8h  (same as A)
DEV int b_dw(int j, int h) { return (j < 4 ? j : j + 4) + 4 * h; }

#if USE_ASYNC
typedef const char __attribute__((address_space(3))) as3_char;
DEV unsigned lds_addr_of(const void* p) {
  return (unsigned)(unsigned long long)(as3_char*)p;
}
// global -> LDS async copy, 16 bytes per lane (ASYNCcnt-tracked)
DEV void async_b128(unsigned lds_byte, const void* gp) {
  asm volatile("global_load_async_to_lds_b128 %0, %1, off"
               :: "v"(lds_byte), "v"((unsigned long long)gp)
               : "memory");
}
DEV void wait_async0() { asm volatile("s_wait_asynccnt 0" ::: "memory"); }
#endif

// ---------------------------------------------------------------------------
// fp32 -> bf16 conversion
// ---------------------------------------------------------------------------
__global__ __launch_bounds__(256) void cvt_f32_bf16(const float* __restrict__ in,
                                                    bf16_t* __restrict__ out,
                                                    size_t n) {
  size_t i = (size_t)blockIdx.x * blockDim.x + threadIdx.x;
  if (i < n) out[i] = (bf16_t)in[i];
}

// ---------------------------------------------------------------------------
// Generic bf16 GEMM: C[M,N] = A[M,K] * B[K,N], 128x128 block tile, 8 waves.
// A,B bf16 row-major; C is f32 or bf16. M,N % 128 == 0, K % 32 == 0.
// ---------------------------------------------------------------------------
constexpr int LAS = 40;   // lA row stride (elems): 80B = 16B-aligned, 20-bank shift
constexpr int LBS = 34;   // lB row stride (elems)

template <bool OUT_F32>
__global__ __launch_bounds__(256) void gemm_bf16(const bf16_t* __restrict__ A,
                                                 const bf16_t* __restrict__ Bm,
                                                 void* __restrict__ C,
                                                 int M, int N, int K) {
  __shared__ bf16_t lA[128 * LAS];  // [m][k]
  __shared__ bf16_t lB[128 * LBS];  // [n][k] (B transposed so K-pairs adjacent)

  const int tid = threadIdx.x;
  const int wid = tid >> 5, lane = tid & 31;
  const int lh = lane >> 4, ln = lane & 15;
  const int wm = wid & 3;   // wave row:  32 rows each (4 waves)
  const int wn = wid >> 2;  // wave col:  64 cols each (2 waves)
  const int bm = blockIdx.y * 128, bn = blockIdx.x * 128;

  const int ar = tid >> 1, acp = (tid & 1) * 16;  // A staging: 32 bytes/thread
  const int bk = tid >> 3, bng = (tid & 7) * 16;  // B staging: 16 elems/thread
#if USE_ASYNC
  const unsigned lA_dst = lds_addr_of(&lA[ar * LAS + acp]);
#endif

  v8f acc[2][4] = {};

  for (int k0 = 0; k0 < K; k0 += 32) {
    __syncthreads();
    {  // stage A tile 128x32
      const bf16_t* g = A + (size_t)(bm + ar) * K + k0 + acp;
#if USE_ASYNC
      async_b128(lA_dst, g);
      async_b128(lA_dst + 16, (const char*)g + 16);
#else
      const unsigned int* gu = (const unsigned int*)g;
      unsigned int* s = (unsigned int*)(&lA[ar * LAS + acp]);
#pragma unroll
      for (int i = 0; i < 8; i++) s[i] = gu[i];
#endif
    }
    {  // stage B tile 32x128 transposed -> lB[n][k]
      const bf16_t* g = Bm + (size_t)(k0 + bk) * N + bn + bng;
#pragma unroll
      for (int i = 0; i < 16; i++) lB[(bng + i) * LBS + bk] = g[i];
    }
    if (k0 + 32 < K) {  // prefetch next tiles into L2
      __builtin_prefetch(A + (size_t)(bm + ar) * K + k0 + 32 + acp, 0, 1);
      __builtin_prefetch(Bm + (size_t)(k0 + 32 + bk) * N + bn + bng, 0, 1);
    }
#if USE_ASYNC
    wait_async0();
#endif
    __syncthreads();

    v16bf afr[2];
#pragma unroll
    for (int mi = 0; mi < 2; mi++) {
      int row = wm * 32 + mi * 16 + ln;
      const unsigned int* s = (const unsigned int*)(&lA[row * LAS]);
      v8u u;
#pragma unroll
      for (int j = 0; j < 8; j++) u[j] = s[a_dw(j, lh)];
      afr[mi] = as_v16bf(u);
    }
#pragma unroll
    for (int ni = 0; ni < 4; ni++) {
      int col = wn * 64 + ni * 16 + ln;
      const unsigned int* s = (const unsigned int*)(&lB[col * LBS]);
      v8u u;
#pragma unroll
      for (int j = 0; j < 8; j++) u[j] = s[b_dw(j, lh)];
      v16bf bfr = as_v16bf(u);
#pragma unroll
      for (int mi = 0; mi < 2; mi++) acc[mi][ni] = wmma_bf16(afr[mi], bfr, acc[mi][ni]);
    }
  }

#pragma unroll
  for (int mi = 0; mi < 2; mi++)
#pragma unroll
    for (int ni = 0; ni < 4; ni++)
#pragma unroll
      for (int r = 0; r < 8; r++) {
        int row = bm + wm * 32 + mi * 16 + r + 8 * lh;
        int col = bn + wn * 64 + ni * 16 + ln;
        float v = acc[mi][ni][r];
        if constexpr (OUT_F32)
          ((float*)C)[(size_t)row * N + col] = v;
        else
          ((bf16_t*)C)[(size_t)row * N + col] = (bf16_t)v;
      }
}

// ---------------------------------------------------------------------------
// Pack q/k token-major: (T,H*64) nope + (T,H*64) pe -> (T,H*128), RoPE on pe,
// optional scale folded in (1/sqrt(HD) for Q).
// ---------------------------------------------------------------------------
__global__ __launch_bounds__(256) void pack_qk_rope(const bf16_t* __restrict__ nope,
                                                    const bf16_t* __restrict__ pe,
                                                    bf16_t* __restrict__ out,
                                                    float mul) {
  size_t idx = (size_t)blockIdx.x * blockDim.x + threadIdx.x;  // t*H*128 + h*128 + r
  if (idx >= (size_t)Tt * Hh * 128) return;
  int r = (int)(idx & 127);
  size_t th = idx >> 7;
  int h = (int)(th % Hh);
  size_t t = th / Hh;
  int s = (int)(t % Ss);
  size_t obase = t * (Hh * 128) + (size_t)h * 128;
  if (r < 64) {
    out[obase + r] = (bf16_t)((float)nope[t * (Hh * 64) + h * 64 + r] * mul);
  } else if (r < 96) {
    int j = r - 64;
    float t1 = (float)pe[t * (Hh * 64) + h * 64 + j];
    float t2 = (float)pe[t * (Hh * 64) + h * 64 + j + 32];
    float invf = __expf(-(float)j * LOGB_OVER_HALF);
    float ang = (float)s * invf;
    float c = __cosf(ang), sn = __sinf(ang);
    out[obase + 64 + j] = (bf16_t)((t1 * c - t2 * sn) * mul);
    out[obase + 96 + j] = (bf16_t)((t1 * sn + t2 * c) * mul);
  }
}

// ---------------------------------------------------------------------------
// Flash attention: block = 4 waves x 16 queries; 32-key blocks staged in LDS.
// Q,K,V,Out all token-major (T, H*128) bf16. Q pre-scaled by 1/sqrt(HD).
// ---------------------------------------------------------------------------
constexpr int LKS = 136;  // lK row stride (elems): 272B, 16B-aligned, 68-bank shift
constexpr int LVS = 34;   // lV row stride
constexpr int LPS = 34;   // lP row stride
constexpr int ROWE = Hh * HDd;  // 2048 elems per token row

__global__ __launch_bounds__(128) void flash_attn(const bf16_t* __restrict__ Q,
                                                  const bf16_t* __restrict__ Kk,
                                                  const bf16_t* __restrict__ V,
                                                  bf16_t* __restrict__ Out) {
  __shared__ bf16_t lK[32 * LKS];       // [key][hd]
  __shared__ bf16_t lV[128 * LVS];      // [hd][key] (transposed)
  __shared__ bf16_t lP[4 * 16 * LPS];   // per-wave P tile [row][key]

  const int bh = blockIdx.x;
  const int b = bh / Hh, h = bh % Hh;
  const int qbase = blockIdx.y * 64;
  const int tid = threadIdx.x, wid = tid >> 5, lane = tid & 31;
  const int lh = lane >> 4, ln = lane & 15;
  const int qw = qbase + wid * 16;
  const size_t tok0 = (size_t)b * Ss;     // first token of this batch
  const size_t hoff = (size_t)h * HDd;    // head offset within a token row

  // preload Q fragments (16 queries x 128 dims = 4 chunks of K=32)
  v16bf qf[4];
  {
    const unsigned int* g =
        (const unsigned int*)(Q + (tok0 + qw + ln) * ROWE + hoff);
#pragma unroll
    for (int c = 0; c < 4; c++) {
      v8u u;
#pragma unroll
      for (int j = 0; j < 8; j++) u[j] = g[c * 16 + a_dw(j, lh)];
      qf[c] = as_v16bf(u);
    }
  }

  v8f O[8] = {};
  float mr[8], lr[8];
#pragma unroll
  for (int r = 0; r < 8; r++) { mr[r] = NEGF; lr[r] = 0.0f; }

  const int sk = tid >> 2, sseg = (tid & 3) * 32;  // staging: key row, 32-elem seg
#if USE_ASYNC
  const unsigned lK_dst = lds_addr_of(&lK[sk * LKS + sseg]);
#endif

  const int nkb = qbase / 32 + 2;  // causal: keys up to qbase+63
  for (int kb = 0; kb < nkb; kb++) {
    __syncthreads();
    {  // stage K block 32x128 -> lK[key][hd]
      const bf16_t* g = Kk + (tok0 + kb * 32 + sk) * ROWE + hoff + sseg;
#if USE_ASYNC
      async_b128(lK_dst,      g);
      async_b128(lK_dst + 16, (const char*)g + 16);
      async_b128(lK_dst + 32, (const char*)g + 32);
      async_b128(lK_dst + 48, (const char*)g + 48);
#else
      const unsigned int* gu = (const unsigned int*)g;
      unsigned int* s = (unsigned int*)(&lK[sk * LKS + sseg]);
#pragma unroll
      for (int i = 0; i < 16; i++) s[i] = gu[i];
#endif
    }
    {  // stage V block 32x128 transposed -> lV[hd][key]
      const bf16_t* g = V + (tok0 + kb * 32 + sk) * ROWE + hoff + sseg;
#pragma unroll
      for (int i = 0; i < 32; i++) lV[(sseg + i) * LVS + sk] = g[i];
    }
#if USE_ASYNC
    wait_async0();
#endif
    __syncthreads();

    // scores S = Q * K^T : two 16x16 tiles (keys 0-15, 16-31), K-dim 128 chained
    v8f s0 = {}, s1 = {};
    {
      const unsigned int* r0 = (const unsigned int*)(&lK[ln * LKS]);
      const unsigned int* r1 = (const unsigned int*)(&lK[(16 + ln) * LKS]);
#pragma unroll
      for (int c = 0; c < 4; c++) {
        v8u u0, u1;
#pragma unroll
        for (int j = 0; j < 8; j++) {
          int idx = c * 16 + b_dw(j, lh);  // hd dword within row
          u0[j] = r0[idx];
          u1[j] = r1[idx];
        }
        s0 = wmma_bf16(qf[c], as_v16bf(u0), s0);
        s1 = wmma_bf16(qf[c], as_v16bf(u1), s1);
      }
    }

    // online softmax (row = r + 8*lh, 16 lanes per row; scale pre-folded into Q)
#pragma unroll
    for (int r = 0; r < 8; r++) {
      int qrow = qw + r + 8 * lh;
      float x0 = s0[r];
      float x1 = s1[r];
      if (kb * 32 + ln > qrow) x0 = NEGF;
      if (kb * 32 + 16 + ln > qrow) x1 = NEGF;
      float vm = fmaxf(x0, x1);
#pragma unroll
      for (int off = 1; off < 16; off <<= 1) vm = fmaxf(vm, __shfl_xor(vm, off, 32));
      float mn = fmaxf(mr[r], vm);
      float al = __expf(mr[r] - mn);
      float p0 = __expf(x0 - mn);
      float p1 = __expf(x1 - mn);
      float rs = p0 + p1;
#pragma unroll
      for (int off = 1; off < 16; off <<= 1) rs += __shfl_xor(rs, off, 32);
      lr[r] = lr[r] * al + rs;
      mr[r] = mn;
#pragma unroll
      for (int on = 0; on < 8; on++) O[on][r] *= al;
      int prow = r + 8 * lh;
      lP[(wid * 16 + prow) * LPS + ln] = (bf16_t)p0;
      lP[(wid * 16 + prow) * LPS + 16 + ln] = (bf16_t)p1;
    }
    asm volatile("s_wait_dscnt 0" ::: "memory");  // same-wave LDS write->read

    // load P as A-fragment (16 x 32 keys)
    v16bf pf;
    {
      const unsigned int* s = (const unsigned int*)(&lP[(wid * 16 + ln) * LPS]);
      v8u u;
#pragma unroll
      for (int j = 0; j < 8; j++) u[j] = s[a_dw(j, lh)];
      pf = as_v16bf(u);
    }
    // O += P * V : 8 WMMAs over HD=128
#pragma unroll
    for (int on = 0; on < 8; on++) {
      const unsigned int* s = (const unsigned int*)(&lV[(on * 16 + ln) * LVS]);
      v8u u;
#pragma unroll
      for (int j = 0; j < 8; j++) u[j] = s[b_dw(j, lh)];
      O[on] = wmma_bf16(pf, as_v16bf(u), O[on]);
    }
  }

  // normalize + store token-major
#pragma unroll
  for (int r = 0; r < 8; r++) {
    float inv = 1.0f / lr[r];
    int row = qw + r + 8 * lh;
#pragma unroll
    for (int on = 0; on < 8; on++) {
      Out[(tok0 + row) * ROWE + hoff + on * 16 + ln] = (bf16_t)(O[on][r] * inv);
    }
  }
}

// ---------------------------------------------------------------------------
// Host orchestration
// ---------------------------------------------------------------------------
extern "C" void kernel_launch(void* const* d_in, const int* in_sizes, int n_in,
                              void* d_out, int out_size, void* d_ws, size_t ws_size,
                              hipStream_t stream) {
  (void)in_sizes; (void)n_in; (void)out_size; (void)ws_size;
  const float* x       = (const float*)d_in[0];
  const float* Wq_down = (const float*)d_in[1];
  const float* Wq_up   = (const float*)d_in[2];
  const float* Wq_rope = (const float*)d_in[3];
  const float* Wkv_down= (const float*)d_in[4];
  const float* Wk_up   = (const float*)d_in[5];
  const float* Wk_rope = (const float*)d_in[6];
  const float* Wv_up   = (const float*)d_in[7];
  const float* Wo      = (const float*)d_in[8];

  char* p = (char*)d_ws;
  auto alloc = [&](size_t nelem) -> bf16_t* {
    bf16_t* r = (bf16_t*)p;
    p += (nelem * sizeof(bf16_t) + 255) & ~(size_t)255;
    return r;
  };

  bf16_t* xb     = alloc((size_t)Tt * Dd);
  bf16_t* w_qd   = alloc((size_t)Dd * QRr);
  bf16_t* w_qu   = alloc((size_t)QRr * Hh * NDd);
  bf16_t* w_qr   = alloc((size_t)QRr * Hh * RDd);
  bf16_t* w_kvd  = alloc((size_t)Dd * KVRr);
  bf16_t* w_ku   = alloc((size_t)KVRr * Hh * NDd);
  bf16_t* w_kr   = alloc((size_t)Dd * Hh * RDd);
  bf16_t* w_vu   = alloc((size_t)KVRr * Hh * HDd);
  bf16_t* w_o    = alloc((size_t)Hh * HDd * Dd);
  bf16_t* q_lat  = alloc((size_t)Tt * QRr);
  bf16_t* kv_lat = alloc((size_t)Tt * KVRr);
  bf16_t* q_nope = alloc((size_t)Tt * Hh * NDd);
  bf16_t* q_pe   = alloc((size_t)Tt * Hh * RDd);
  bf16_t* k_nope = alloc((size_t)Tt * Hh * NDd);
  bf16_t* k_pe   = alloc((size_t)Tt * Hh * RDd);
  bf16_t* v_rm   = alloc((size_t)Tt * Hh * HDd);   // token-major V (used directly)
  bf16_t* Qt     = alloc((size_t)Tt * Hh * HDd);   // token-major Q (scaled, roped)
  bf16_t* Kt     = alloc((size_t)Tt * Hh * HDd);   // token-major K (roped)
  bf16_t* attn   = alloc((size_t)Tt * Hh * HDd);

  auto cvt = [&](const float* src, bf16_t* dst, size_t n) {
    cvt_f32_bf16<<<dim3((unsigned)((n + 255) / 256)), dim3(256), 0, stream>>>(src, dst, n);
  };
  cvt(x, xb, (size_t)Tt * Dd);
  cvt(Wq_down, w_qd, (size_t)Dd * QRr);
  cvt(Wq_up, w_qu, (size_t)QRr * Hh * NDd);
  cvt(Wq_rope, w_qr, (size_t)QRr * Hh * RDd);
  cvt(Wkv_down, w_kvd, (size_t)Dd * KVRr);
  cvt(Wk_up, w_ku, (size_t)KVRr * Hh * NDd);
  cvt(Wk_rope, w_kr, (size_t)Dd * Hh * RDd);
  cvt(Wv_up, w_vu, (size_t)KVRr * Hh * HDd);
  cvt(Wo, w_o, (size_t)Hh * HDd * Dd);

  auto gemm = [&](const bf16_t* A, const bf16_t* Bm, void* C, int M, int N, int K,
                  bool f32out) {
    dim3 g(N / 128, M / 128);
    if (f32out)
      gemm_bf16<true><<<g, dim3(256), 0, stream>>>(A, Bm, C, M, N, K);
    else
      gemm_bf16<false><<<g, dim3(256), 0, stream>>>(A, Bm, C, M, N, K);
  };

  gemm(xb, w_qd, q_lat, Tt, QRr, Dd, false);            // q_lat  = x @ Wq_down
  gemm(q_lat, w_qu, q_nope, Tt, Hh * NDd, QRr, false);  // q_nope = q_lat @ Wq_up
  gemm(q_lat, w_qr, q_pe, Tt, Hh * RDd, QRr, false);    // q_pe   = q_lat @ Wq_rope
  gemm(xb, w_kvd, kv_lat, Tt, KVRr, Dd, false);         // kv_lat = x @ Wkv_down
  gemm(kv_lat, w_ku, k_nope, Tt, Hh * NDd, KVRr, false);
  gemm(xb, w_kr, k_pe, Tt, Hh * RDd, Dd, false);        // k_pe   = x @ Wk_rope
  gemm(kv_lat, w_vu, v_rm, Tt, Hh * HDd, KVRr, false);  // v      = kv_lat @ Wv_up

  {
    size_t n = (size_t)Tt * Hh * 128;
    dim3 g((unsigned)((n + 255) / 256));
    pack_qk_rope<<<g, dim3(256), 0, stream>>>(q_nope, q_pe, Qt, SCALE);
    pack_qk_rope<<<g, dim3(256), 0, stream>>>(k_nope, k_pe, Kt, 1.0f);
  }

  flash_attn<<<dim3(Bb * Hh, Ss / 64), dim3(128), 0, stream>>>(Qt, Kt, v_rm, attn);

  gemm(attn, w_o, d_out, Tt, Dd, Hh * HDd, true);       // out = attn @ Wo (f32)
}
